// Head_81011673137482
// MI455X (gfx1250) — matrix operand order repaired and checked by reading, not compiled, not run on hardware
//
#include <hip/hip_runtime.h>
#include <hip/hip_bf16.h>

typedef __attribute__((ext_vector_type(16))) _Float16 v16h;
typedef __attribute__((ext_vector_type(8)))  _Float16 v8h;
typedef __attribute__((ext_vector_type(8)))  float    v8f;
typedef __attribute__((ext_vector_type(4)))  float    v4f;

#define HS 128      // head size
#define NE 1024     // embed dim
#define TT 2048     // sequence length
#define NB 4        // batch

static __device__ __forceinline__ v8f wmma_f16(v16h a, v16h b, v8f c) {
    return __builtin_amdgcn_wmma_f32_16x16x32_f16(false, a, false, b, (short)0, c,
                                                  false, false);
}

// ---------------------------------------------------------------------------
// Kernel 1: convert + transpose weights to f16, Wt[m][h][c], m: 0=Wq(scaled),
// 1=Wk, 2=Wv. Folding 1/sqrt(HS) into Wq folds the softmax scale into Q.
// ---------------------------------------------------------------------------
__global__ void convert_w(const float* __restrict__ Wk,
                          const float* __restrict__ Wq,
                          const float* __restrict__ Wv,
                          _Float16* __restrict__ Wt) {
    int idx = blockIdx.x * blockDim.x + threadIdx.x;
    if (idx >= 3 * HS * NE) return;
    int m   = idx / (HS * NE);
    int rem = idx - m * (HS * NE);
    int h   = rem / NE;
    int c   = rem - h * NE;
    const float* src = (m == 0) ? Wq : (m == 1) ? Wk : Wv;
    float v = src[c * HS + h];
    if (m == 0) v *= 0.08838834764831845f;  // HS^-0.5
    Wt[idx] = (_Float16)v;
}

// ---------------------------------------------------------------------------
// Kernel 2: fused QKV projection. One wave per (16-row strip, 16-col h-tile),
// computing Q, K, V tiles together so the x A-operand (and its f32->f16
// conversion) is loaded once and feeds three WMMAs per k-step.
// grid = B*T/16; block = 256 (8 waves -> 8 h-tiles of one row strip).
// Q,K stored row-major f16 [b][t][h]; V stored transposed f16 [b][h][t].
// ---------------------------------------------------------------------------
__global__ __launch_bounds__(256) void qkv_proj(const float* __restrict__ x,
                                                const _Float16* __restrict__ Wt,
                                                _Float16* __restrict__ Qh,
                                                _Float16* __restrict__ Kh,
                                                _Float16* __restrict__ Vt) {
    const int lane  = threadIdx.x & 31;
    const int w     = threadIdx.x >> 5;
    const int strip = blockIdx.x;            // 0..B*T/16-1
    const int b     = strip >> 7;            // 128 strips per batch
    const int t0    = (strip & 127) << 4;
    const int h0    = w << 4;
    const int n     = lane & 15;
    const int hi    = lane >> 4;
    const int rbase = hi << 3;
    const int aoff  = hi ? 8 : 0;            // f32 A-run base (per ISA A layout)
    const int boff  = hi ? 8 : 0;            // f16 B-run base

    const float*    xp = x  + ((size_t)(b * TT + t0 + n)) * NE;
    const _Float16* wq = Wt + ((size_t)0 * HS + h0 + n) * NE;
    const _Float16* wk = Wt + ((size_t)1 * HS + h0 + n) * NE;
    const _Float16* wv = Wt + ((size_t)2 * HS + h0 + n) * NE;

    v8f aq = {}, ak = {}, av = {};
    for (int c0 = 0; c0 < NE; c0 += 32) {
        // A: 16x32 chunk of x, f32 -> f16. Lane row = n; two 8-elem runs.
        v4f a0 = *(const v4f*)(xp + c0 + aoff);
        v4f a1 = *(const v4f*)(xp + c0 + aoff + 4);
        v4f a2 = *(const v4f*)(xp + c0 + aoff + 16);
        v4f a3 = *(const v4f*)(xp + c0 + aoff + 20);
        __builtin_prefetch(xp + c0 + 32, 0, 1);   // global_prefetch_b8
        v16h A;
        #pragma unroll
        for (int i = 0; i < 4; ++i) {
            A[i]      = (_Float16)a0[i];
            A[i + 4]  = (_Float16)a1[i];
            A[i + 8]  = (_Float16)a2[i];
            A[i + 12] = (_Float16)a3[i];
        }
        // B operands: contiguous runs of Wt rows (c-major) for each matrix.
        v16h Bq, Bk, Bv;
        ((v8h*)&Bq)[0] = *(const v8h*)(wq + c0 + boff);
        ((v8h*)&Bq)[1] = *(const v8h*)(wq + c0 + boff + 16);
        ((v8h*)&Bk)[0] = *(const v8h*)(wk + c0 + boff);
        ((v8h*)&Bk)[1] = *(const v8h*)(wk + c0 + boff + 16);
        ((v8h*)&Bv)[0] = *(const v8h*)(wv + c0 + boff);
        ((v8h*)&Bv)[1] = *(const v8h*)(wv + c0 + boff + 16);
        aq = wmma_f16(A, Bq, aq);
        ak = wmma_f16(A, Bk, ak);
        av = wmma_f16(A, Bv, av);
    }

    // Q, K row-major scalar stores.
    {
        _Float16* dq = Qh + ((size_t)b * TT + t0 + rbase) * HS + h0 + n;
        _Float16* dk = Kh + ((size_t)b * TT + t0 + rbase) * HS + h0 + n;
        #pragma unroll
        for (int r = 0; r < 8; ++r) {
            dq[(size_t)r * HS] = (_Float16)aq[r];
            dk[(size_t)r * HS] = (_Float16)ak[r];
        }
    }
    // V transposed: lane holds column h = h0+n, 8 contiguous t values.
    {
        v8h pk;
        #pragma unroll
        for (int r = 0; r < 8; ++r) pk[r] = (_Float16)av[r];
        _Float16* dst = Vt + ((size_t)b * HS + h0 + n) * TT + t0 + rbase;
        *(v8h*)dst = pk;
    }
}

// ---------------------------------------------------------------------------
// Kernel 3: causal flash attention. One wave per 32-query tile (2 A row-tiles
// -> 4 interleaved S accumulators hide WMMA latency; K/V loads amortized 2x),
// key blocks of 32 rows. Online softmax, P through LDS (layout transpose),
// O += P V via WMMAs over the 8 H-tiles.
// grid = B*T/32/2; block = 64 (2 independent waves).
// ---------------------------------------------------------------------------
__global__ __launch_bounds__(64) void flash_attn(const _Float16* __restrict__ Qh,
                                                 const _Float16* __restrict__ Kh,
                                                 const _Float16* __restrict__ Vt,
                                                 float* __restrict__ out) {
    __shared__ __align__(16) _Float16 lds[2][32 * 32];

    const int lane  = threadIdx.x & 31;
    const int w     = threadIdx.x >> 5;
    const int q     = blockIdx.x * 2 + w;    // 32-row query tile id 0..255
    const int b     = q >> 6;                // 64 tiles per batch
    const int t0    = (q & 63) << 5;
    const int n     = lane & 15;
    const int hi    = lane >> 4;
    const int rbase = hi << 3;

    const _Float16* Qb = Qh + (size_t)b * TT * HS;
    const _Float16* Kb = Kh + (size_t)b * TT * HS;
    const _Float16* Vb = Vt + (size_t)b * HS * TT;

    // Preload the whole 32x128 Q tile as 2x4 A operands; reused all loop.
    v16h Aq[2][4];
    #pragma unroll
    for (int rt = 0; rt < 2; ++rt) {
        const _Float16* qp = Qb + (size_t)(t0 + rt * 16 + n) * HS + (hi ? 8 : 0);
        #pragma unroll
        for (int kc = 0; kc < 4; ++kc) {
            ((v8h*)&Aq[rt][kc])[0] = *(const v8h*)(qp + kc * 32);
            ((v8h*)&Aq[rt][kc])[1] = *(const v8h*)(qp + kc * 32 + 16);
        }
    }

    v8f O[2][8];
    float mx[2][8], ls[2][8];
    #pragma unroll
    for (int rt = 0; rt < 2; ++rt)
        #pragma unroll
        for (int i = 0; i < 8; ++i) {
            O[rt][i] = (v8f){};
            mx[rt][i] = -1e30f;
            ls[rt][i] = 0.0f;
        }

    for (int tk = 0; tk < t0 + 32; tk += 32) {
        // ---- S = Q K^T: 32 queries x 32 keys, four 16x16 C tiles ----
        v8f S[2][2];
        S[0][0] = (v8f){}; S[0][1] = (v8f){};
        S[1][0] = (v8f){}; S[1][1] = (v8f){};
        const _Float16* kp0 = Kb + (size_t)(tk + n) * HS + (hi ? 8 : 0);
        const _Float16* kp1 = kp0 + (size_t)16 * HS;
        __builtin_prefetch(kp0 + (size_t)32 * HS, 0, 1);  // next key block
        #pragma unroll
        for (int kc = 0; kc < 4; ++kc) {
            v16h B0, B1;
            ((v8h*)&B0)[0] = *(const v8h*)(kp0 + kc * 32);
            ((v8h*)&B0)[1] = *(const v8h*)(kp0 + kc * 32 + 16);
            ((v8h*)&B1)[0] = *(const v8h*)(kp1 + kc * 32);
            ((v8h*)&B1)[1] = *(const v8h*)(kp1 + kc * 32 + 16);
            S[0][0] = wmma_f16(Aq[0][kc], B0, S[0][0]);
            S[1][0] = wmma_f16(Aq[1][kc], B0, S[1][0]);
            S[0][1] = wmma_f16(Aq[0][kc], B1, S[0][1]);
            S[1][1] = wmma_f16(Aq[1][kc], B1, S[1][1]);
        }

        // ---- causal mask (only on diagonal-edge blocks) ----
        if (tk + 31 > t0) {
            #pragma unroll
            for (int rt = 0; rt < 2; ++rt)
                #pragma unroll
                for (int r = 0; r < 8; ++r) {
                    int trow = t0 + rt * 16 + rbase + r;
                    if (tk + n > trow)      S[rt][0][r] = -1e30f;
                    if (tk + 16 + n > trow) S[rt][1][r] = -1e30f;
                }
        }

        // ---- online softmax (row stats per C-layout register row) ----
        float corr[2][8];
        #pragma unroll
        for (int rt = 0; rt < 2; ++rt)
            #pragma unroll
            for (int r = 0; r < 8; ++r) {
                float v = fmaxf(S[rt][0][r], S[rt][1][r]);
                v = fmaxf(v, __shfl_xor(v, 1, 32));
                v = fmaxf(v, __shfl_xor(v, 2, 32));
                v = fmaxf(v, __shfl_xor(v, 4, 32));
                v = fmaxf(v, __shfl_xor(v, 8, 32));
                float mnew = fmaxf(mx[rt][r], v);
                corr[rt][r] = __expf(mx[rt][r] - mnew);
                mx[rt][r] = mnew;
                float p0 = __expf(S[rt][0][r] - mnew);
                float p1 = __expf(S[rt][1][r] - mnew);
                S[rt][0][r] = p0; S[rt][1][r] = p1;
                float rs = p0 + p1;
                rs += __shfl_xor(rs, 1, 32);
                rs += __shfl_xor(rs, 2, 32);
                rs += __shfl_xor(rs, 4, 32);
                rs += __shfl_xor(rs, 8, 32);
                ls[rt][r] = ls[rt][r] * corr[rt][r] + rs;
            }
        #pragma unroll
        for (int rt = 0; rt < 2; ++rt)
            #pragma unroll
            for (int ht = 0; ht < 8; ++ht)
                #pragma unroll
                for (int r = 0; r < 8; ++r) O[rt][ht][r] *= corr[rt][r];

        // ---- P: C-layout f32 -> A-layout f16 via LDS (per-wave region) ----
        _Float16* pl = &lds[w][0];
        #pragma unroll
        for (int rt = 0; rt < 2; ++rt)
            #pragma unroll
            for (int r = 0; r < 8; ++r) {
                pl[(rt * 16 + rbase + r) * 32 + n]      = (_Float16)S[rt][0][r];
                pl[(rt * 16 + rbase + r) * 32 + 16 + n] = (_Float16)S[rt][1][r];
            }
        asm volatile("s_wait_dscnt 0" ::: "memory");  // DS is in-order per wave
        v16h Pa[2];
        #pragma unroll
        for (int rt = 0; rt < 2; ++rt) {
            const _Float16* pr = pl + (rt * 16 + n) * 32 + (hi ? 8 : 0);
            ((v8h*)&Pa[rt])[0] = *(const v8h*)(pr);
            ((v8h*)&Pa[rt])[1] = *(const v8h*)(pr + 16);
        }
        asm volatile("s_wait_dscnt 0" ::: "memory");

        // ---- O += P V; each V B-operand feeds both row-tiles ----
        #pragma unroll
        for (int ht = 0; ht < 8; ++ht) {
            const _Float16* vp = Vb + (size_t)(ht * 16 + n) * TT + tk + (hi ? 8 : 0);
            v16h Bv;
            ((v8h*)&Bv)[0] = *(const v8h*)(vp);
            ((v8h*)&Bv)[1] = *(const v8h*)(vp + 16);
            O[0][ht] = wmma_f16(Pa[0], Bv, O[0][ht]);
            O[1][ht] = wmma_f16(Pa[1], Bv, O[1][ht]);
        }
    }

    // ---- normalize and store fp32 output [b][t][h] ----
    #pragma unroll
    for (int rt = 0; rt < 2; ++rt) {
        float* op = out + ((size_t)(b * TT + t0 + rt * 16 + rbase)) * HS;
        #pragma unroll
        for (int r = 0; r < 8; ++r) {
            float inv = 1.0f / ls[rt][r];
            #pragma unroll
            for (int ht = 0; ht < 8; ++ht)
                op[(size_t)r * HS + ht * 16 + n] = O[rt][ht][r] * inv;
        }
    }
}

// ---------------------------------------------------------------------------
extern "C" void kernel_launch(void* const* d_in, const int* in_sizes, int n_in,
                              void* d_out, int out_size, void* d_ws, size_t ws_size,
                              hipStream_t stream) {
    const float* x  = (const float*)d_in[0];
    const float* Wk = (const float*)d_in[1];
    const float* Wq = (const float*)d_in[2];
    const float* Wv = (const float*)d_in[3];
    float* out = (float*)d_out;

    // workspace: Qh(2MB) | Kh(2MB) | Vt(2MB) | Wt(768KB)
    char* ws = (char*)d_ws;
    _Float16* Qh = (_Float16*)(ws);
    _Float16* Kh = (_Float16*)(ws + (size_t)2 * 1024 * 1024);
    _Float16* Vt = (_Float16*)(ws + (size_t)4 * 1024 * 1024);
    _Float16* Wt = (_Float16*)(ws + (size_t)6 * 1024 * 1024);

    convert_w<<<dim3((3 * HS * NE + 255) / 256), dim3(256), 0, stream>>>(Wk, Wq, Wv, Wt);
    qkv_proj<<<dim3(NB * TT / 16), dim3(256), 0, stream>>>(x, Wt, Qh, Kh, Vt);
    flash_attn<<<dim3(NB * TT / 32 / 2), dim3(64), 0, stream>>>(Qh, Kh, Vt, out);
}
